// JointConvLayer_63608465654110
// MI455X (gfx1250) — compile-verified, matrix operand order checked by reading, not run on hardware
//
#include <hip/hip_runtime.h>
#include <math.h>

// ---------------------------------------------------------------------------
// MI455X (gfx1250) implementation.
// Heavy GEMMs (ipmp_msg 126 GF, ipmp_z 34 GF, i2s, ipmp_out) run on
// v_wmma_f32_16x16x32_bf16 (f32 accumulate, bf16 operands). Irregular
// per-edge tensor products run in VALU with weights staged in LDS.
// Segment means via f32 global atomics into zeroed workspace accumulators.
// Weight streams get unconditional speculative global_prefetch_b8 hints
// (prefetch past end-of-buffer is silently dropped; weights sit in L2).
// ---------------------------------------------------------------------------

typedef __bf16 bf16_t;
typedef __attribute__((ext_vector_type(16))) __bf16 v16bf;
typedef __attribute__((ext_vector_type(8)))  __bf16 v8bf;
typedef __attribute__((ext_vector_type(8)))  float  v8f;

#define N_ATOM 40000
#define N_RES  4000
#define E_BOND 80000
#define E_RAD  640000
#define E_ATOM (E_BOND + E_RAD)
#define E_RES  128000
#define C0A 64
#define C1A 16
#define DA  112
#define C0R 128
#define C1R 32
#define DR  224
#define CS  384
#define CZ  128
#define HE  64
#define MSG_K  900
#define MSG_KP 928   // pad 900 -> 29*32
#define ZK     896   // 28*32 exactly

// ---------------- WMMA helpers (CDNA5 16x16x32 bf16) -----------------------
// A fragment (16x32 tile, row-major, `stride` elems/row):
//   lane m = lane&15 ; chunks K=[koff,koff+8) and K=[koff+16,koff+24),
//   koff = (lane>>4)*8  (per ISA 16-bit A layout).
// B fragment loaded identically from W^T (N x K row-major).
static __device__ __forceinline__ v16bf frag_load(const bf16_t* base, int stride, int lane) {
  int m    = lane & 15;
  int koff = (lane >> 4) << 3;
  const bf16_t* p = base + (size_t)m * stride + koff;
  union { v16bf v; v8bf h[2]; } u;
  u.h[0] = *reinterpret_cast<const v8bf*>(p);
  u.h[1] = *reinterpret_cast<const v8bf*>(p + 16);
  return u.v;
}

static __device__ __forceinline__ v8f wmma_bf16(v16bf a, v16bf b, v8f c) {
  return __builtin_amdgcn_wmma_f32_16x16x32_bf16(false, a, false, b, (short)0, c, false, false);
}

static __device__ __forceinline__ v8f vzero8() {
  v8f z = {0.f,0.f,0.f,0.f,0.f,0.f,0.f,0.f};
  return z;
}

// ---------------- utility kernels ------------------------------------------
__global__ void k_zero(float* p, int n) {
  int i = blockIdx.x * 256 + threadIdx.x;
  if (i < n) p[i] = 0.f;
}

__global__ void k_f2bf(bf16_t* d, const float* s, int n) {
  int i = blockIdx.x * 256 + threadIdx.x;
  if (i < n) d[i] = (bf16_t)s[i];
}

// W (K x N, row-major, f32) -> W^T (N x Kpad, row-major, bf16, zero padded K)
__global__ void k_wt(bf16_t* d, const float* s, int K, int N, int Kp) {
  int i = blockIdx.x * 256 + threadIdx.x;
  if (i >= N * Kp) return;
  int n = i / Kp, k = i - n * Kp;
  d[i] = (k < K) ? (bf16_t)s[(size_t)k * N + n] : (bf16_t)0.f;
}

__global__ void k_count(float* cnt, const int* idx, int n) {
  int i = blockIdx.x * 256 + threadIdx.x;
  if (i < n) atomicAdd(&cnt[idx[i]], 1.f);
}

// ---------------- gate MLP (64 -> 64 relu -> 5), VALU, weights in LDS ------
__global__ void __launch_bounds__(256) k_gate(const float* x, const float* w1, const float* b1,
                                              const float* w2, const float* b2, float* g, int n) {
  __shared__ float sw1[HE * HE];
  __shared__ float sb1[HE];
  __shared__ float sw2[HE * 5];
  __shared__ float sb2[5];
  for (int i = threadIdx.x; i < HE * HE; i += 256) sw1[i] = w1[i];
  for (int i = threadIdx.x; i < HE; i += 256)      sb1[i] = b1[i];
  for (int i = threadIdx.x; i < HE * 5; i += 256)  sw2[i] = w2[i];
  if (threadIdx.x < 5) sb2[threadIdx.x] = b2[threadIdx.x];
  __syncthreads();
  int e = blockIdx.x * 256 + threadIdx.x;
  if (e >= n) return;
  float xr[HE];
  for (int c = 0; c < HE; c++) xr[c] = x[(size_t)e * HE + c];
  float out[5];
  for (int o = 0; o < 5; o++) out[o] = sb2[o];
  for (int j = 0; j < HE; j++) {
    float a = sb1[j];
    for (int c = 0; c < HE; c++) a += xr[c] * sw1[c * HE + j];
    a = fmaxf(a, 0.f);
    for (int o = 0; o < 5; o++) out[o] += a * sw2[j * 5 + o];
  }
  for (int o = 0; o < 5; o++) g[(size_t)e * 5 + o] = out[o];
}

// ---------------- tensor-product edge message (VALU, LDS weights) ----------
// out_s = g0*sh0*(s@W1) + g1*((v.sh1)@W2)
// out_v[f] = g2*(s@W3)[f]*sh1 + g3*sh0*(sum_c v[c]*W4[c,f]) + g4*(sum_c (v[c] x sh1)*W5[c,f])
// NOTE: counts for the segment mean come ONLY from k_count (no count here).
template <int C0O, int C1O>
__global__ void __launch_bounds__(256) k_tp(const float* feats, const int* srcI, const int* dstI,
                                            const float* sh, const float* g,
                                            const float* W1, const float* W2, const float* W3,
                                            const float* W4, const float* W5,
                                            float* accum, int nE) {
  __shared__ float sW1[C0A * C0O];
  __shared__ float sW2[C1A * C0O];
  __shared__ float sW3[C0A * C1O];
  __shared__ float sW4[C1A * C1O];
  __shared__ float sW5[C1A * C1O];
  for (int i = threadIdx.x; i < C0A * C0O; i += 256) sW1[i] = W1[i];
  for (int i = threadIdx.x; i < C1A * C0O; i += 256) sW2[i] = W2[i];
  for (int i = threadIdx.x; i < C0A * C1O; i += 256) sW3[i] = W3[i];
  for (int i = threadIdx.x; i < C1A * C1O; i += 256) sW4[i] = W4[i];
  for (int i = threadIdx.x; i < C1A * C1O; i += 256) sW5[i] = W5[i];
  __syncthreads();
  int e = blockIdx.x * 256 + threadIdx.x;
  if (e >= nE) return;
  int a   = srcI ? srcI[e] : e;
  int dTo = dstI[e];
  const float* f = feats + (size_t)a * DA;
  float s[C0A], vx[C1A], vy[C1A], vz[C1A], sv[C1A];
  for (int c = 0; c < C0A; c++) s[c] = f[c];
  float h1x = sh[(size_t)e * 4 + 1], h1y = sh[(size_t)e * 4 + 2], h1z = sh[(size_t)e * 4 + 3];
  float sh0 = sh[(size_t)e * 4 + 0];
  for (int c = 0; c < C1A; c++) {
    vx[c] = f[C0A + c * 3 + 0];
    vy[c] = f[C0A + c * 3 + 1];
    vz[c] = f[C0A + c * 3 + 2];
    sv[c] = vx[c] * h1x + vy[c] * h1y + vz[c] * h1z;
  }
  const float* ge = g + (size_t)e * 5;
  float g0 = ge[0], g1 = ge[1], g2 = ge[2], g3 = ge[3], g4 = ge[4];
  const int DOUT = C0O + 3 * C1O;
  float* acc_row = accum + (size_t)dTo * DOUT;
  float f0 = g0 * sh0, g3s = g3 * sh0;
  for (int fo = 0; fo < C0O; fo++) {
    float a1 = 0.f, a2 = 0.f;
    for (int c = 0; c < C0A; c++) a1 += s[c] * sW1[c * C0O + fo];
    for (int c = 0; c < C1A; c++) a2 += sv[c] * sW2[c * C0O + fo];
    atomicAdd(&acc_row[fo], f0 * a1 + g1 * a2);
  }
  for (int fo = 0; fo < C1O; fo++) {
    float p = 0.f;
    for (int c = 0; c < C0A; c++) p += s[c] * sW3[c * C1O + fo];
    p *= g2;
    float ox = p * h1x, oy = p * h1y, oz = p * h1z;
    float bx = 0.f, by = 0.f, bz = 0.f, cx = 0.f, cy = 0.f, cz = 0.f;
    for (int c = 0; c < C1A; c++) {
      float w4 = sW4[c * C1O + fo];
      bx += vx[c] * w4; by += vy[c] * w4; bz += vz[c] * w4;
      float w5 = sW5[c * C1O + fo];
      cx += (vy[c] * h1z - vz[c] * h1y) * w5;
      cy += (vz[c] * h1x - vx[c] * h1z) * w5;
      cz += (vx[c] * h1y - vy[c] * h1x) * w5;
    }
    ox += g3s * bx + g4 * cx;
    oy += g3s * by + g4 * cy;
    oz += g3s * bz + g4 * cz;
    atomicAdd(&acc_row[C0O + fo * 3 + 0], ox);
    atomicAdd(&acc_row[C0O + fo * 3 + 1], oy);
    atomicAdd(&acc_row[C0O + fo * 3 + 2], oz);
  }
}

// atom_feats = atom_features + accum/max(cnt,1)  (writes d_out region 0)
__global__ void k_atom_final(const float* in, const float* accum, const float* cnt,
                             float* out, int ntot) {
  int i = blockIdx.x * 256 + threadIdx.x;
  if (i >= ntot) return;
  int r = i / DA;
  out[i] = in[i] + accum[i] / fmaxf(cnt[r], 1.f);
}

// quats -> R (stored), res_update = mean + rotate vector part, -> bf16
__global__ void k_resupd(const float* accum, const float* cnt, const float* quats,
                         float* Rout, bf16_t* ru_bf) {
  int r = blockIdx.x * 64 + threadIdx.x;
  if (r >= N_RES) return;
  float qw = quats[r * 4 + 0], qx = quats[r * 4 + 1], qy = quats[r * 4 + 2], qz = quats[r * 4 + 3];
  float inv_n = rsqrtf(qw * qw + qx * qx + qy * qy + qz * qz);
  qw *= inv_n; qx *= inv_n; qy *= inv_n; qz *= inv_n;
  float R[9];
  R[0] = 1.f - 2.f * (qy * qy + qz * qz); R[1] = 2.f * (qx * qy - qw * qz); R[2] = 2.f * (qx * qz + qw * qy);
  R[3] = 2.f * (qx * qy + qw * qz); R[4] = 1.f - 2.f * (qx * qx + qz * qz); R[5] = 2.f * (qy * qz - qw * qx);
  R[6] = 2.f * (qx * qz - qw * qy); R[7] = 2.f * (qy * qz + qw * qx); R[8] = 1.f - 2.f * (qx * qx + qy * qy);
  for (int k = 0; k < 9; k++) Rout[r * 9 + k] = R[k];
  float inv = 1.f / fmaxf(cnt[r], 1.f);
  const float* ar = accum + (size_t)r * DR;
  bf16_t* orow = ru_bf + (size_t)r * DR;
  for (int c = 0; c < C0R; c++) orow[c] = (bf16_t)(ar[c] * inv);
  for (int c = 0; c < C1R; c++) {
    float x = ar[C0R + c * 3 + 0] * inv;
    float y = ar[C0R + c * 3 + 1] * inv;
    float z = ar[C0R + c * 3 + 2] * inv;
    // out[i] = sum_j R[j][i] * v[j]  (R^T v)
    orow[C0R + c * 3 + 0] = (bf16_t)(R[0] * x + R[3] * y + R[6] * z);
    orow[C0R + c * 3 + 1] = (bf16_t)(R[1] * x + R[4] * y + R[7] * z);
    orow[C0R + c * 3 + 2] = (bf16_t)(R[2] * x + R[5] * y + R[8] * z);
  }
}

// ---------------- generic WMMA GEMM: out = act(A @ W^T' + bias [+ residual])
// grid = (M/16, N/64), block = 128 (4 waves, one 16x16 n-tile each)
__global__ void __launch_bounds__(128) k_gemm(const bf16_t* A, int lda,
                                              const bf16_t* WT, int ldw,
                                              const float* bias, const float* residual,
                                              float* outF, bf16_t* outB, int ldo,
                                              int K, int act) {
  int lane = threadIdx.x & 31, wave = threadIdx.x >> 5;
  int rowBase = blockIdx.x * 16;
  int colBase = blockIdx.y * 64 + wave * 16;
  v8f acc = vzero8();
  const bf16_t* Ab = A + (size_t)rowBase * lda;
  const bf16_t* Wb = WT + (size_t)colBase * ldw;
  for (int k = 0; k < K; k += 32) {
    // speculative prefetch of next k-chunk; past-the-end is silently dropped
    __builtin_prefetch((const void*)(Wb + k + 32), 0, 3);
    v16bf af = frag_load(Ab + k, lda, lane);
    v16bf bfg = frag_load(Wb + k, ldw, lane);
    acc = wmma_bf16(af, bfg, acc);
  }
  int col = colBase + (lane & 15);
  float bv = bias ? bias[col] : 0.f;
  int rh = (lane >> 4) << 3;
  for (int v = 0; v < 8; v++) {
    int row = rowBase + v + rh;
    float x = acc[v] + bv;
    if (residual) x += residual[(size_t)row * ldo + col];
    if (act) x = fmaxf(x, 0.f);
    if (outF) outF[(size_t)row * ldo + col] = x;
    if (outB) outB[(size_t)row * ldo + col] = (bf16_t)x;
  }
}

// ---------------- layernorm (one block per row, C <= 384) -------------------
__global__ void __launch_bounds__(128) k_ln(const float* x, const float* addend,
                                            const float* gamma, const float* beta,
                                            const float* mask,
                                            float* outF, bf16_t* outB, int C) {
  __shared__ float row[CS];
  __shared__ float red[128];
  int r = blockIdx.x, t = threadIdx.x;
  float lsum = 0.f;
  for (int c = t; c < C; c += 128) {
    float v = x[(size_t)r * C + c];
    if (addend) v += addend[(size_t)r * C + c];
    row[c] = v;
    lsum += v;
  }
  red[t] = lsum;
  __syncthreads();
  for (int s2 = 64; s2 > 0; s2 >>= 1) { if (t < s2) red[t] += red[t + s2]; __syncthreads(); }
  float mean = red[0] / C;
  __syncthreads();
  float lvar = 0.f;
  for (int c = t; c < C; c += 128) { float dv = row[c] - mean; lvar += dv * dv; }
  red[t] = lvar;
  __syncthreads();
  for (int s2 = 64; s2 > 0; s2 >>= 1) { if (t < s2) red[t] += red[t + s2]; __syncthreads(); }
  float rstd = rsqrtf(red[0] / C + 1e-5f);
  float mk = mask ? mask[r] : 1.f;
  for (int c = t; c < C; c += 128) {
    float y = ((row[c] - mean) * rstd * gamma[c] + beta[c]) * mk;
    if (outF) outF[(size_t)r * C + c] = y;
    if (outB) outB[(size_t)r * C + c] = (bf16_t)y;
  }
}

// agg_mean -> bf16
__global__ void k_aggnorm(const float* acc, const float* cnt, bf16_t* out, int C, int n) {
  int i = blockIdx.x * 256 + threadIdx.x;
  if (i >= n) return;
  int r = i / C;
  out[i] = (bf16_t)(acc[i] / fmaxf(cnt[r], 1.f));
}

// ---------------- fused ipmp_msg: gather -> 900->384 relu -> 384->384 -> atomic scatter
__global__ void __launch_bounds__(128) k_ipmp_msg(const bf16_t* s_res_bf, const bf16_t* ref_bf,
                                                  const float* trans, const float* Rw,
                                                  const int* ei, const int* ej,
                                                  const bf16_t* W1T, const float* b1,
                                                  const bf16_t* W2T, const float* b2,
                                                  float* agg_accum) {
  __shared__ __align__(16) bf16_t At[16][MSG_KP];
  __shared__ __align__(16) bf16_t Ht[16][CS];
  __shared__ int sTarget[16];
  int tid = threadIdx.x;
  int eBase = blockIdx.x * 16;
  if (tid < 16) {
    int e = eBase + tid;
    int i = ei[e], j = ej[e];
    sTarget[tid] = i;
    float dx = trans[j * 3 + 0] - trans[i * 3 + 0];
    float dy = trans[j * 3 + 1] - trans[i * 3 + 1];
    float dz = trans[j * 3 + 2] - trans[i * 3 + 2];
    const float* R = Rw + (size_t)i * 9;  // d_loc = R^T d
    float d0 = R[0] * dx + R[3] * dy + R[6] * dz;
    float d1 = R[1] * dx + R[4] * dy + R[7] * dz;
    float d2 = R[2] * dx + R[5] * dy + R[8] * dz;
    float dn = sqrtf(d0 * d0 + d1 * d1 + d2 * d2);
    At[tid][896] = (bf16_t)d0; At[tid][897] = (bf16_t)d1;
    At[tid][898] = (bf16_t)d2; At[tid][899] = (bf16_t)dn;
    for (int k = MSG_K; k < MSG_KP; k++) At[tid][k] = (bf16_t)0.f;
  }
  for (int idx = tid; idx < 16 * 896; idx += 128) {
    int r = idx / 896, c = idx - r * 896;
    int e = eBase + r;
    bf16_t v;
    if (c < CS)            v = s_res_bf[(size_t)ei[e] * CS + c];
    else if (c < 2 * CS)   v = s_res_bf[(size_t)ej[e] * CS + (c - CS)];
    else                   v = ref_bf[(size_t)e * CZ + (c - 2 * CS)];
    At[r][c] = v;
  }
  __syncthreads();
  int lane = tid & 31, wave = tid >> 5;
  v8f acc[6];
  for (int t = 0; t < 6; t++) acc[t] = vzero8();
  for (int kc = 0; kc < MSG_KP; kc += 32) {
    v16bf a = frag_load(&At[0][kc], MSG_KP, lane);
#pragma unroll
    for (int t = 0; t < 6; t++) {
      int n0 = wave * 96 + t * 16;
      __builtin_prefetch((const void*)(W1T + (size_t)n0 * MSG_KP + kc + 32), 0, 3);
      v16bf b = frag_load(W1T + (size_t)n0 * MSG_KP + kc, MSG_KP, lane);
      acc[t] = wmma_bf16(a, b, acc[t]);
    }
  }
  int rh = (lane >> 4) << 3;
  for (int t = 0; t < 6; t++) {
    int col = wave * 96 + t * 16 + (lane & 15);
    float bv = b1[col];
    for (int v = 0; v < 8; v++) Ht[v + rh][col] = (bf16_t)fmaxf(acc[t][v] + bv, 0.f);
  }
  __syncthreads();
  v8f acc2[6];
  for (int t = 0; t < 6; t++) acc2[t] = vzero8();
  for (int kc = 0; kc < CS; kc += 32) {
    v16bf a = frag_load(&Ht[0][kc], CS, lane);
#pragma unroll
    for (int t = 0; t < 6; t++) {
      int n0 = wave * 96 + t * 16;
      v16bf b = frag_load(W2T + (size_t)n0 * CS + kc, CS, lane);
      acc2[t] = wmma_bf16(a, b, acc2[t]);
    }
  }
  for (int t = 0; t < 6; t++) {
    int col = wave * 96 + t * 16 + (lane & 15);
    float bv = b2[col];
    for (int v = 0; v < 8; v++) {
      int row = v + rh;
      atomicAdd(&agg_accum[(size_t)sTarget[row] * CS + col], acc2[t][v] + bv);
    }
  }
}

// ---------------- fused ipmp_z: gather -> 896->128 relu -> 128->128 + residual
__global__ void __launch_bounds__(128) k_ipmp_z(const bf16_t* s_out_bf, const bf16_t* ref_bf,
                                                const float* ref_f32,
                                                const int* ei, const int* ej,
                                                const bf16_t* W1T, const float* b1,
                                                const bf16_t* W2T, const float* b2,
                                                float* z_out) {
  __shared__ __align__(16) bf16_t At[16][ZK];
  __shared__ __align__(16) bf16_t Ht[16][CZ];
  int tid = threadIdx.x;
  int eBase = blockIdx.x * 16;
  for (int idx = tid; idx < 16 * ZK; idx += 128) {
    int r = idx / ZK, c = idx - r * ZK;
    int e = eBase + r;
    bf16_t v;
    if (c < CS)          v = s_out_bf[(size_t)ei[e] * CS + c];
    else if (c < 2 * CS) v = s_out_bf[(size_t)ej[e] * CS + (c - CS)];
    else                 v = ref_bf[(size_t)e * CZ + (c - 2 * CS)];
    At[r][c] = v;
  }
  __syncthreads();
  int lane = tid & 31, wave = tid >> 5;
  v8f acc[2];
  acc[0] = vzero8(); acc[1] = vzero8();
  for (int kc = 0; kc < ZK; kc += 32) {
    v16bf a = frag_load(&At[0][kc], ZK, lane);
#pragma unroll
    for (int t = 0; t < 2; t++) {
      int n0 = wave * 32 + t * 16;
      __builtin_prefetch((const void*)(W1T + (size_t)n0 * ZK + kc + 32), 0, 3);
      v16bf b = frag_load(W1T + (size_t)n0 * ZK + kc, ZK, lane);
      acc[t] = wmma_bf16(a, b, acc[t]);
    }
  }
  int rh = (lane >> 4) << 3;
  for (int t = 0; t < 2; t++) {
    int col = wave * 32 + t * 16 + (lane & 15);
    float bv = b1[col];
    for (int v = 0; v < 8; v++) Ht[v + rh][col] = (bf16_t)fmaxf(acc[t][v] + bv, 0.f);
  }
  __syncthreads();
  v8f acc2[2];
  acc2[0] = vzero8(); acc2[1] = vzero8();
  for (int kc = 0; kc < CZ; kc += 32) {
    v16bf a = frag_load(&Ht[0][kc], CZ, lane);
#pragma unroll
    for (int t = 0; t < 2; t++) {
      int n0 = wave * 32 + t * 16;
      v16bf b = frag_load(W2T + (size_t)n0 * CZ + kc, CZ, lane);
      acc2[t] = wmma_bf16(a, b, acc2[t]);
    }
  }
  for (int t = 0; t < 2; t++) {
    int col = wave * 32 + t * 16 + (lane & 15);
    float bv = b2[col];
    for (int v = 0; v < 8; v++) {
      int e = eBase + v + rh;
      z_out[(size_t)e * CZ + col] = ref_f32[(size_t)e * CZ + col] + acc2[t][v] + bv;
    }
  }
}

// ---------------------------------------------------------------------------
// Input index map (setup_inputs dict order; `params` pytree flattened with
// JAX sorted-key convention):
//  0 atom_features  1 bond_features  2 radius_edge_features  3 atom_edge_sh
//  4 res_features   5 agg_edge_features  6 agg_edge_sh  7 rigid_quats
//  8 rigid_trans    9 res_edge_features 10 res_mask
// 11-14 gate_agg{b1,b2,w1,w2} 15-18 gate_bond{...} 19-22 gate_rad{...}
// 23-28 i2s{b1,b2,b3,w1,w2,w3} 29-32 ipmp_msg{b1,b2,w1,w2}
// 33-34 ipmp_out{b,w} 35-38 ipmp_z{b1,b2,w1,w2}
// 39-40 ln1{b,g} 41-42 ln2{b,g}
// 43-47 tp_agg{W1..W5} 48-52 tp_atom{W1..W5}
// 53 atom_edge_index(2xE) 54 atom_res_batch 55 res_edge_index(2xE_RES)
// ---------------------------------------------------------------------------
extern "C" void kernel_launch(void* const* d_in, const int* in_sizes, int n_in,
                              void* d_out, int out_size, void* d_ws, size_t ws_size,
                              hipStream_t stream) {
  (void)in_sizes; (void)n_in; (void)out_size; (void)ws_size;
  const float* atom_features = (const float*)d_in[0];
  const float* bond_features = (const float*)d_in[1];
  const float* rad_features  = (const float*)d_in[2];
  const float* atom_edge_sh  = (const float*)d_in[3];
  const float* res_features  = (const float*)d_in[4];
  const float* agg_features  = (const float*)d_in[5];
  const float* agg_sh        = (const float*)d_in[6];
  const float* rigid_quats   = (const float*)d_in[7];
  const float* rigid_trans   = (const float*)d_in[8];
  const float* res_edge_feat = (const float*)d_in[9];
  const float* res_mask      = (const float*)d_in[10];

  const float* ga_b1 = (const float*)d_in[11];
  const float* ga_b2 = (const float*)d_in[12];
  const float* ga_w1 = (const float*)d_in[13];
  const float* ga_w2 = (const float*)d_in[14];
  const float* gb_b1 = (const float*)d_in[15];
  const float* gb_b2 = (const float*)d_in[16];
  const float* gb_w1 = (const float*)d_in[17];
  const float* gb_w2 = (const float*)d_in[18];
  const float* gr_b1 = (const float*)d_in[19];
  const float* gr_b2 = (const float*)d_in[20];
  const float* gr_w1 = (const float*)d_in[21];
  const float* gr_w2 = (const float*)d_in[22];
  const float* i2s_b1 = (const float*)d_in[23];
  const float* i2s_b2 = (const float*)d_in[24];
  const float* i2s_b3 = (const float*)d_in[25];
  const float* i2s_w1 = (const float*)d_in[26];
  const float* i2s_w2 = (const float*)d_in[27];
  const float* i2s_w3 = (const float*)d_in[28];
  const float* msg_b1 = (const float*)d_in[29];
  const float* msg_b2 = (const float*)d_in[30];
  const float* msg_w1 = (const float*)d_in[31];
  const float* msg_w2 = (const float*)d_in[32];
  const float* out_b  = (const float*)d_in[33];
  const float* out_w  = (const float*)d_in[34];
  const float* z_b1   = (const float*)d_in[35];
  const float* z_b2   = (const float*)d_in[36];
  const float* z_w1   = (const float*)d_in[37];
  const float* z_w2   = (const float*)d_in[38];
  const float* ln1_b  = (const float*)d_in[39];
  const float* ln1_g  = (const float*)d_in[40];
  const float* ln2_b  = (const float*)d_in[41];
  const float* ln2_g  = (const float*)d_in[42];
  const float* tA_W1  = (const float*)d_in[43];
  const float* tA_W2  = (const float*)d_in[44];
  const float* tA_W3  = (const float*)d_in[45];
  const float* tA_W4  = (const float*)d_in[46];
  const float* tA_W5  = (const float*)d_in[47];
  const float* tB_W1  = (const float*)d_in[48];
  const float* tB_W2  = (const float*)d_in[49];
  const float* tB_W3  = (const float*)d_in[50];
  const float* tB_W4  = (const float*)d_in[51];
  const float* tB_W5  = (const float*)d_in[52];

  const int* aei   = (const int*)d_in[53];
  const int* srcA  = aei;            // atom_edge_index[0]
  const int* dstA  = aei + E_ATOM;   // atom_edge_index[1]
  const int* batch = (const int*)d_in[54];
  const int* rei   = (const int*)d_in[55];
  const int* ej    = rei;            // res_edge_index[0]
  const int* ei    = rei + E_RES;    // res_edge_index[1]

  float* out_atom = (float*)d_out;
  float* out_sres = out_atom + (size_t)N_ATOM * DA;
  float* out_z    = out_sres + (size_t)N_RES * CS;

  size_t off = 0;
  auto alloc = [&](size_t bytes) -> void* {
    void* p = (char*)d_ws + off;
    off += (bytes + 255) & ~(size_t)255;
    return p;
  };
  bf16_t* wt_i2s1 = (bf16_t*)alloc((size_t)CS * DR * 2);
  bf16_t* wt_i2s2 = (bf16_t*)alloc((size_t)CS * CS * 2);
  bf16_t* wt_i2s3 = (bf16_t*)alloc((size_t)CS * CS * 2);
  bf16_t* wt_msg1 = (bf16_t*)alloc((size_t)CS * MSG_KP * 2);
  bf16_t* wt_msg2 = (bf16_t*)alloc((size_t)CS * CS * 2);
  bf16_t* wt_out  = (bf16_t*)alloc((size_t)CS * CS * 2);
  bf16_t* wt_z1   = (bf16_t*)alloc((size_t)CZ * ZK * 2);
  bf16_t* wt_z2   = (bf16_t*)alloc((size_t)CZ * CZ * 2);
  bf16_t* ref_bf  = (bf16_t*)alloc((size_t)E_RES * CZ * 2);
  float* g_atom   = (float*)alloc((size_t)E_ATOM * 5 * 4);
  float* g_agg    = (float*)alloc((size_t)N_ATOM * 5 * 4);
  float* atom_acc = (float*)alloc((size_t)N_ATOM * DA * 4);
  float* atom_cnt = (float*)alloc((size_t)N_ATOM * 4);
  float* res_acc  = (float*)alloc((size_t)N_RES * DR * 4);
  float* res_cnt  = (float*)alloc((size_t)N_RES * 4);
  float* Rmat     = (float*)alloc((size_t)N_RES * 9 * 4);
  bf16_t* ru_bf   = (bf16_t*)alloc((size_t)N_RES * DR * 2);
  bf16_t* h1_bf   = (bf16_t*)alloc((size_t)N_RES * CS * 2);
  bf16_t* h2_bf   = (bf16_t*)alloc((size_t)N_RES * CS * 2);
  float* i2s_out  = (float*)alloc((size_t)N_RES * CS * 4);
  float* s_res_f  = (float*)alloc((size_t)N_RES * CS * 4);
  bf16_t* s_res_bf= (bf16_t*)alloc((size_t)N_RES * CS * 2);
  float* agg_acc  = (float*)alloc((size_t)N_RES * CS * 4);
  float* ei_cnt   = (float*)alloc((size_t)N_RES * 4);
  bf16_t* agg_bf  = (bf16_t*)alloc((size_t)N_RES * CS * 2);
  float* mid      = (float*)alloc((size_t)N_RES * CS * 4);
  bf16_t* s_out_bf= (bf16_t*)alloc((size_t)N_RES * CS * 2);

  auto g1 = [](int n) { return dim3((unsigned)((n + 255) / 256)); };

  // 1. weight conversions (transpose to NxKpad bf16)
  k_wt<<<g1(CS * DR), 256, 0, stream>>>(wt_i2s1, i2s_w1, DR, CS, DR);
  k_wt<<<g1(CS * CS), 256, 0, stream>>>(wt_i2s2, i2s_w2, CS, CS, CS);
  k_wt<<<g1(CS * CS), 256, 0, stream>>>(wt_i2s3, i2s_w3, CS, CS, CS);
  k_wt<<<g1(CS * MSG_KP), 256, 0, stream>>>(wt_msg1, msg_w1, MSG_K, CS, MSG_KP);
  k_wt<<<g1(CS * CS), 256, 0, stream>>>(wt_msg2, msg_w2, CS, CS, CS);
  k_wt<<<g1(CS * CS), 256, 0, stream>>>(wt_out, out_w, CS, CS, CS);
  k_wt<<<g1(CZ * ZK), 256, 0, stream>>>(wt_z1, z_w1, ZK, CZ, ZK);
  k_wt<<<g1(CZ * CZ), 256, 0, stream>>>(wt_z2, z_w2, CZ, CZ, CZ);
  k_f2bf<<<g1(E_RES * CZ), 256, 0, stream>>>(ref_bf, res_edge_feat, E_RES * CZ);

  // 2. zero accumulators
  k_zero<<<g1(N_ATOM * DA), 256, 0, stream>>>(atom_acc, N_ATOM * DA);
  k_zero<<<g1(N_ATOM), 256, 0, stream>>>(atom_cnt, N_ATOM);
  k_zero<<<g1(N_RES * DR), 256, 0, stream>>>(res_acc, N_RES * DR);
  k_zero<<<g1(N_RES), 256, 0, stream>>>(res_cnt, N_RES);
  k_zero<<<g1(N_RES * CS), 256, 0, stream>>>(agg_acc, N_RES * CS);
  k_zero<<<g1(N_RES), 256, 0, stream>>>(ei_cnt, N_RES);

  // 3. gate MLPs
  k_gate<<<g1(E_BOND), 256, 0, stream>>>(bond_features, gb_w1, gb_b1, gb_w2, gb_b2, g_atom, E_BOND);
  k_gate<<<g1(E_RAD), 256, 0, stream>>>(rad_features, gr_w1, gr_b1, gr_w2, gr_b2,
                                        g_atom + (size_t)E_BOND * 5, E_RAD);
  k_gate<<<g1(N_ATOM), 256, 0, stream>>>(agg_features, ga_w1, ga_b1, ga_w2, ga_b2, g_agg, N_ATOM);

  // 4. atom-level tensor-product messages + seg mean -> atom_feats (d_out)
  k_count<<<g1(E_ATOM), 256, 0, stream>>>(atom_cnt, dstA, E_ATOM);
  k_tp<C0A, C1A><<<g1(E_ATOM), 256, 0, stream>>>(atom_features, srcA, dstA, atom_edge_sh, g_atom,
                                                 tB_W1, tB_W2, tB_W3, tB_W4, tB_W5,
                                                 atom_acc, E_ATOM);
  k_atom_final<<<g1(N_ATOM * DA), 256, 0, stream>>>(atom_features, atom_acc, atom_cnt,
                                                    out_atom, N_ATOM * DA);

  // 5. atom->res aggregation tensor product
  k_count<<<g1(N_ATOM), 256, 0, stream>>>(res_cnt, batch, N_ATOM);
  k_tp<C0R, C1R><<<g1(N_ATOM), 256, 0, stream>>>(out_atom, (const int*)nullptr, batch, agg_sh, g_agg,
                                                 tA_W1, tA_W2, tA_W3, tA_W4, tA_W5,
                                                 res_acc, N_ATOM);

  // 6. res_update (rotate) -> bf16 ; i2s MLP (WMMA) ; ln1
  k_resupd<<<dim3((N_RES + 63) / 64), 64, 0, stream>>>(res_acc, res_cnt, rigid_quats, Rmat, ru_bf);
  k_gemm<<<dim3(N_RES / 16, CS / 64), 128, 0, stream>>>(ru_bf, DR, wt_i2s1, DR, i2s_b1, nullptr,
                                                        nullptr, h1_bf, CS, DR, 1);
  k_gemm<<<dim3(N_RES / 16, CS / 64), 128, 0, stream>>>(h1_bf, CS, wt_i2s2, CS, i2s_b2, nullptr,
                                                        nullptr, h2_bf, CS, CS, 1);
  k_gemm<<<dim3(N_RES / 16, CS / 64), 128, 0, stream>>>(h2_bf, CS, wt_i2s3, CS, i2s_b3, nullptr,
                                                        i2s_out, nullptr, CS, CS, 0);
  k_ln<<<dim3(N_RES), 128, 0, stream>>>(i2s_out, res_features, ln1_g, ln1_b, nullptr,
                                        s_res_f, s_res_bf, CS);

  // 7. ipmp message passing (fused WMMA) + seg mean + ipmp_out + ln2
  k_count<<<g1(E_RES), 256, 0, stream>>>(ei_cnt, ei, E_RES);
  k_ipmp_msg<<<dim3(E_RES / 16), 128, 0, stream>>>(s_res_bf, ref_bf, rigid_trans, Rmat, ei, ej,
                                                   wt_msg1, msg_b1, wt_msg2, msg_b2, agg_acc);
  k_aggnorm<<<g1(N_RES * CS), 256, 0, stream>>>(agg_acc, ei_cnt, agg_bf, CS, N_RES * CS);
  k_gemm<<<dim3(N_RES / 16, CS / 64), 128, 0, stream>>>(agg_bf, CS, wt_out, CS, out_b, s_res_f,
                                                        mid, nullptr, CS, CS, 0);
  k_ln<<<dim3(N_RES), 128, 0, stream>>>(mid, nullptr, ln2_g, ln2_b, res_mask,
                                        out_sres, s_out_bf, CS);

  // 8. ipmp_z (fused WMMA) -> z_out
  k_ipmp_z<<<dim3(E_RES / 16), 128, 0, stream>>>(s_out_bf, ref_bf, res_edge_feat, ei, ej,
                                                 wt_z1, z_b1, wt_z2, z_b2, out_z);
}
